// GraphSageEncoder_41755672051882
// MI455X (gfx1250) — matrix operand order repaired and checked by reading, not compiled, not run on hardware
//
#include <hip/hip_runtime.h>

// ---------------- problem constants ----------------
#define G_   32
#define N_   1024
#define E_   16384
#define FIN  128
#define HID  1024
#define OUT_ 4096
#define BB   32
#define SS   256

typedef __attribute__((ext_vector_type(16))) __bf16 v16bf;
typedef __attribute__((ext_vector_type(8)))  __bf16 v8bf;
typedef __attribute__((ext_vector_type(8)))  float  v8f;

// ---------------- workspace layout (bytes, all 256-aligned) ----------------
// zeroed region first: agg1 | cnt | sumH | sumA2  (contiguous floats)
#define OFF_AGG1  ((size_t)0)                         // G*N*FIN f32   = 16 MB
#define OFF_CNT   ((size_t)16777216)                  // G*N f32
#define OFF_SUMH  ((size_t)16908288)                  // G*HID f32
#define OFF_SUMA2 ((size_t)17039360)                  // G*HID f32
#define ZERO_FLOATS 4292608                           // (above four) / 4
#define OFF_XCAT  ((size_t)17170432)                  // G*N*256 bf16  = 16 MB
#define OFF_W1T   ((size_t)33947648)                  // 1024*256 bf16
#define OFF_H     ((size_t)34471936)                  // G*N*HID bf16  = 64 MB
#define OFF_W2T   ((size_t)101580800)                 // 4096*2048 bf16 = 16 MB
#define OFF_A2    ((size_t)118358016)                 // 32*2048 bf16
#define OFF_GEMB  ((size_t)118489088)                 // 32*4096 f32
#define OFF_NID   ((size_t)119013376)                 // BB*SS int32

// ---------------- helpers ----------------
__global__ void k_zero(float* p, int n) {
  int i = blockIdx.x * blockDim.x + threadIdx.x;
  int st = gridDim.x * blockDim.x;
  for (; i < n; i += st) p[i] = 0.f;
}

// W1t[n][k] bf16, k<128 -> Wl1[k][n], else Wr1[k-128][n]
__global__ void k_prepw1(const float* __restrict__ Wl1, const float* __restrict__ Wr1,
                         __bf16* __restrict__ W1t) {
  int i = blockIdx.x * blockDim.x + threadIdx.x;
  if (i >= 1024 * 256) return;
  int n = i >> 8, k = i & 255;
  float v = (k < 128) ? Wl1[k * 1024 + n] : Wr1[(k - 128) * 1024 + n];
  W1t[i] = (__bf16)v;
}

// W2t[n][k] bf16, k<1024 -> Wl2[k][n], else Wr2[k-1024][n]
__global__ void k_prepw2(const float* __restrict__ Wl2, const float* __restrict__ Wr2,
                         __bf16* __restrict__ W2t) {
  int i = blockIdx.x * blockDim.x + threadIdx.x;
  if (i >= 4096 * 2048) return;
  int n = i >> 11, k = i & 2047;
  float v = (k < 1024) ? Wl2[(size_t)k * 4096 + n] : Wr2[(size_t)(k - 1024) * 4096 + n];
  W2t[i] = (__bf16)v;
}

// layer-1 scatter aggregation: one block per (g,e), 128 threads = features
__global__ void k_agg1(const float* __restrict__ x, const int* __restrict__ ei,
                       float* __restrict__ agg1, float* __restrict__ cnt) {
  int blk = blockIdx.x;
  int g = blk / E_;
  int e = blk - g * E_;
  const int* eg = ei + (size_t)g * 2 * E_;
  int s = eg[e];
  int d = eg[E_ + e];
  int f = threadIdx.x;
  atomicAdd(&agg1[((size_t)g * N_ + d) * FIN + f], x[((size_t)g * N_ + s) * FIN + f]);
  if (f == 0) atomicAdd(&cnt[g * N_ + d], 1.0f);
}

// build Xcat = [agg1/max(cnt,1) | x] in bf16, row-major K=256
__global__ void k_xcat(const float* __restrict__ x, const float* __restrict__ agg1,
                       const float* __restrict__ cnt, __bf16* __restrict__ Xcat) {
  size_t gn = blockIdx.x;           // g*N + n
  int k = threadIdx.x;              // 0..127
  float c = cnt[gn];
  float inv = 1.0f / (c > 1.0f ? c : 1.0f);
  Xcat[gn * 256 + k]       = (__bf16)(agg1[gn * FIN + k] * inv);
  Xcat[gn * 256 + 128 + k] = (__bf16)(x[gn * FIN + k]);
}

// ---- WMMA fragment loaders (bf16 16x16x32, contiguous k = 16*(lane/16) + s in A and B) ----
__device__ __forceinline__ v16bf load_a_frag(const __bf16* A, int ld, int m0, int k0, int lane) {
  int r  = lane & 15;
  int kh = (lane < 16) ? 0 : 16;
  const __bf16* p = A + (size_t)(m0 + r) * ld + (k0 + kh);
  union { v16bf v; v8bf h[2]; } u;
  u.h[0] = *(const v8bf*)(p);        // K = k0+kh   .. +7
  u.h[1] = *(const v8bf*)(p + 8);    // K = k0+kh+8 .. +15
  return u.v;
}
__device__ __forceinline__ v16bf load_b_frag(const __bf16* Bt, int ld, int n0, int k0, int lane) {
  int c  = lane & 15;
  int kh = (lane < 16) ? 0 : 16;
  const __bf16* p = Bt + (size_t)(n0 + c) * ld + (k0 + kh);  // 16 contiguous bf16
  return *(const v16bf*)p;
}

// GEMM1: per graph  h = relu(Xcat[1024,256] @ W1tᵀ + b1) -> bf16 [1024,1024]
// block = 128 thr (4 waves), tile 64x64; wave -> 16(M) x 64(N)
__global__ void __launch_bounds__(128) k_gemm1(const __bf16* __restrict__ Xcat,
                                               const __bf16* __restrict__ W1t,
                                               const float* __restrict__ b1,
                                               __bf16* __restrict__ h) {
  int bx = blockIdx.x;              // G * 16 * 16
  int g  = bx >> 8;
  int mb = (bx >> 4) & 15;
  int nb = bx & 15;
  int wave = threadIdx.x >> 5;
  int lane = threadIdx.x & 31;
  int m0 = mb * 64 + wave * 16;
  int n0 = nb * 64;
  const __bf16* Ag = Xcat + (size_t)g * N_ * 256;

  v8f acc[4];
#pragma unroll
  for (int j = 0; j < 4; ++j)
#pragma unroll
    for (int r = 0; r < 8; ++r) acc[j][r] = 0.f;

  for (int k0 = 0; k0 < 256; k0 += 32) {
    v16bf a = load_a_frag(Ag, 256, m0, k0, lane);
#pragma unroll
    for (int j = 0; j < 4; ++j) {
      v16bf b = load_b_frag(W1t, 256, n0 + j * 16, k0, lane);
      acc[j] = __builtin_amdgcn_wmma_f32_16x16x32_bf16(false, a, false, b,
                                                       (short)0, acc[j], false, false);
    }
  }

  __bf16* hg = h + (size_t)g * N_ * HID;
  int mo = (lane < 16) ? 0 : 8;
  int cl = lane & 15;
#pragma unroll
  for (int j = 0; j < 4; ++j) {
    int nc = n0 + j * 16 + cl;
    float bias = b1[nc];
#pragma unroll
    for (int r = 0; r < 8; ++r) {
      float v = acc[j][r] + bias;
      v = v > 0.f ? v : 0.f;
      hg[(size_t)(m0 + mo + r) * HID + nc] = (__bf16)v;
    }
  }
}

// column sums of h per graph -> sumH[g][n]
__global__ void k_colmean(const __bf16* __restrict__ h, float* __restrict__ sumH) {
  int g  = blockIdx.x >> 2;
  int nc = ((blockIdx.x & 3) << 8) + threadIdx.x;   // 256 thr
  const __bf16* hg = h + (size_t)g * N_ * HID;
  float s = 0.f;
  for (int m = 0; m < N_; ++m) s += (float)hg[(size_t)m * HID + nc];
  sumH[g * HID + nc] = s;
}

// weighted edge-sum: sumA2[g][f] += sum_e h[src][f] / max(cnt[dst],1)
__global__ void __launch_bounds__(1024) k_agg2(const __bf16* __restrict__ h,
                                               const int* __restrict__ ei,
                                               const float* __restrict__ cnt,
                                               float* __restrict__ sumA2) {
  int g  = blockIdx.x >> 3;
  int ch = blockIdx.x & 7;
  int f  = threadIdx.x;                  // 1024 features
  const int* eg = ei + (size_t)g * 2 * E_;
  const __bf16* hg = h + (size_t)g * N_ * HID;
  const float* cg = cnt + g * N_;
  float s = 0.f;
  int e0 = ch * (E_ / 8);
  for (int e = e0; e < e0 + E_ / 8; ++e) {
    int src = eg[e];
    int dst = eg[E_ + e];
    float c = cg[dst];
    float w = 1.f / (c > 1.f ? c : 1.f);
    s += w * (float)hg[(size_t)src * HID + f];
  }
  atomicAdd(&sumA2[g * HID + f], s);
}

// A2 = bf16([sumA2 | sumH] / N)   -> [32, 2048]
__global__ void k_a2(const float* __restrict__ sumA2, const float* __restrict__ sumH,
                     __bf16* __restrict__ A2) {
  int i = blockIdx.x * blockDim.x + threadIdx.x;
  if (i >= G_ * 2048) return;
  int g = i >> 11, k = i & 2047;
  float v = (k < 1024) ? sumA2[g * HID + k] : sumH[g * HID + (k - 1024)];
  A2[i] = (__bf16)(v * (1.0f / (float)N_));
}

// GEMM2: graph_emb[32,4096] = A2[32,2048] @ W2tᵀ + b2   (one 16x16 tile per wave)
__global__ void __launch_bounds__(128) k_gemm2(const __bf16* __restrict__ A2,
                                               const __bf16* __restrict__ W2t,
                                               const float* __restrict__ b2,
                                               float* __restrict__ gemb) {
  int wave = threadIdx.x >> 5;
  int lane = threadIdx.x & 31;
  int t  = blockIdx.x * 4 + wave;   // 0..511
  int mt = t & 1;
  int nt = t >> 1;
  int m0 = mt * 16, n0 = nt * 16;
  v8f acc;
#pragma unroll
  for (int r = 0; r < 8; ++r) acc[r] = 0.f;
  for (int k0 = 0; k0 < 2048; k0 += 32) {
    v16bf a = load_a_frag(A2, 2048, m0, k0, lane);
    v16bf b = load_b_frag(W2t, 2048, n0, k0, lane);
    acc = __builtin_amdgcn_wmma_f32_16x16x32_bf16(false, a, false, b,
                                                  (short)0, acc, false, false);
  }
  int mo = (lane < 16) ? 0 : 8;
  int nc = n0 + (lane & 15);
  float bias = b2[nc];
#pragma unroll
  for (int r = 0; r < 8; ++r)
    gemb[(size_t)(m0 + mo + r) * OUT_ + nc] = acc[r] + bias;
}

// serial prefix over is_node -> nid[i] = clipped graph index or -1
__global__ void k_order(const unsigned char* __restrict__ is_node, int* __restrict__ nid) {
  if (threadIdx.x != 0 || blockIdx.x != 0) return;
  int c = 0;
  for (int i = 0; i < BB * SS; ++i) {
    if (is_node[i]) { nid[i] = (c < G_) ? c : (G_ - 1); ++c; }
    else nid[i] = -1;
  }
}

// final gather: each block copies one 4096-f32 row
__global__ void __launch_bounds__(256) k_out(const int* __restrict__ nid,
                                             const int* __restrict__ ids,
                                             const float* __restrict__ gemb,
                                             const float* __restrict__ emb,
                                             float* __restrict__ out) {
  int row = blockIdx.x;
  int t = threadIdx.x;
  int k = nid[row];
  const float4* src = (k >= 0) ? (const float4*)(gemb + (size_t)k * OUT_)
                               : (const float4*)(emb + (size_t)ids[row] * OUT_);
  float4* dst = (float4*)(out + (size_t)row * OUT_);
#pragma unroll
  for (int j = 0; j < 4; ++j) dst[t + j * 256] = src[t + j * 256];
}

extern "C" void kernel_launch(void* const* d_in, const int* in_sizes, int n_in,
                              void* d_out, int out_size, void* d_ws, size_t ws_size,
                              hipStream_t stream) {
  (void)in_sizes; (void)n_in; (void)out_size; (void)ws_size;
  const float* x   = (const float*)d_in[0];
  const int*   ei  = (const int*)d_in[1];
  const int*   ids = (const int*)d_in[2];
  const unsigned char* isn = (const unsigned char*)d_in[3];
  const float* Wl1 = (const float*)d_in[4];
  const float* Wr1 = (const float*)d_in[5];
  const float* b1  = (const float*)d_in[6];
  const float* Wl2 = (const float*)d_in[7];
  const float* Wr2 = (const float*)d_in[8];
  const float* b2  = (const float*)d_in[9];
  const float* emb = (const float*)d_in[10];
  float* out = (float*)d_out;

  char* ws = (char*)d_ws;
  float*  agg1  = (float*)(ws + OFF_AGG1);
  float*  cnt   = (float*)(ws + OFF_CNT);
  float*  sumH  = (float*)(ws + OFF_SUMH);
  float*  sumA2 = (float*)(ws + OFF_SUMA2);
  __bf16* Xcat  = (__bf16*)(ws + OFF_XCAT);
  __bf16* W1t   = (__bf16*)(ws + OFF_W1T);
  __bf16* h     = (__bf16*)(ws + OFF_H);
  __bf16* W2t   = (__bf16*)(ws + OFF_W2T);
  __bf16* A2    = (__bf16*)(ws + OFF_A2);
  float*  gemb  = (float*)(ws + OFF_GEMB);
  int*    nid   = (int*)(ws + OFF_NID);

  k_zero  <<<4096, 256, 0, stream>>>(agg1, ZERO_FLOATS);
  k_prepw1<<<1024, 256, 0, stream>>>(Wl1, Wr1, W1t);
  k_prepw2<<<32768, 256, 0, stream>>>(Wl2, Wr2, W2t);
  k_agg1  <<<G_ * E_, FIN, 0, stream>>>(x, ei, agg1, cnt);
  k_xcat  <<<G_ * N_, FIN, 0, stream>>>(x, agg1, cnt, Xcat);
  k_gemm1 <<<G_ * 16 * 16, 128, 0, stream>>>(Xcat, W1t, b1, h);
  k_colmean<<<G_ * 4, 256, 0, stream>>>(h, sumH);
  k_agg2  <<<G_ * 8, 1024, 0, stream>>>(h, ei, cnt, sumA2);
  k_a2    <<<(G_ * 2048 + 255) / 256, 256, 0, stream>>>(sumA2, sumH, A2);
  k_gemm2 <<<128, 128, 0, stream>>>(A2, W2t, b2, gemb);
  k_order <<<1, 1, 0, stream>>>(isn, nid);
  k_out   <<<BB * SS, 256, 0, stream>>>(nid, ids, gemb, emb, out);
}
